// GraphConditioning_88811333747253
// MI455X (gfx1250) — matrix operand order repaired and checked by reading, not compiled, not run on hardware
//
#include <hip/hip_runtime.h>

// ---------------------------------------------------------------------------
// MI455X (gfx1250, wave32) implementation of the hetero-GNN forward pass.
// All dense layers run through v_wmma_f32_16x16x32_bf16 with fp32 accumulate;
// weights are pre-transposed to bf16 so A and B fragments are contiguous
// global_load_b128s. Subtile count is a template parameter so the K-loop has
// no control flow: all fragment loads issue as one clause, then NT WMMAs.
// Activations load / fp32 stores use non-temporal hints (streamed data);
// weights keep default RT policy (reused by every block, WGP$-resident).
// Scatter (segment_sum) uses global_atomic_add_f32. Workspace: ~513 MB.
// ---------------------------------------------------------------------------

typedef __attribute__((ext_vector_type(16))) __bf16 v16bf;
typedef __attribute__((ext_vector_type(8)))  float  v8f;
typedef unsigned int u32x4 __attribute__((ext_vector_type(4)));

#define NU 100000
#define NI 50000
#define NEDGE 600000

__device__ __forceinline__ __bf16 f2bf(float f) {
  union { float f; unsigned u; } x; x.f = f;
  unsigned r = x.u + 0x7FFFu + ((x.u >> 16) & 1u);   // round-to-nearest-even
  union { unsigned short s; __bf16 b; } y;
  y.s = (unsigned short)(r >> 16);
  return y.b;
}

// ---------------- elementwise helpers -------------------------------------
__global__ void k_zero(float* __restrict__ p, long n) {
  long i = (long)blockIdx.x * 256 + threadIdx.x;
  if (i < n) p[i] = 0.f;
}

__global__ void k_convert_b16(const float* __restrict__ a, __bf16* __restrict__ o, long n) {
  long i = (long)blockIdx.x * 256 + threadIdx.x;
  if (i < n) o[i] = f2bf(a[i]);
}

__global__ void k_add_convert_b16(const float* __restrict__ a, const float* __restrict__ b,
                                  __bf16* __restrict__ o, long n) {
  long i = (long)blockIdx.x * 256 + threadIdx.x;
  if (i < n) o[i] = f2bf(a[i] + b[i]);
}

// Wt[n*K + k] = bf16(W[k*N + n])   (W is [K,N] row-major)
__global__ void k_transpose_b16(const float* __restrict__ W, __bf16* __restrict__ Wt,
                                int K, int N) {
  long i = (long)blockIdx.x * 256 + threadIdx.x;
  if (i >= (long)K * N) return;
  int k = (int)(i / N), n = (int)(i % N);
  Wt[(size_t)n * K + k] = f2bf(W[i]);
}

// EDM positional embedding -> bf16 [n,128]
__global__ void k_posemb(const float* __restrict__ pos, __bf16* __restrict__ o, int n) {
  long id = (long)blockIdx.x * 256 + threadIdx.x;
  if (id >= (long)n * 128) return;
  int r = (int)(id >> 7), c = (int)(id & 127), i = c & 63;
  float f = __powf(1.0f / 2000.0f, (float)i * (1.0f / 64.0f));
  float x = pos[r] * f;
  o[id] = f2bf(c < 64 ? __cosf(x) : __sinf(x));
}

// m[dst] += h[src] over edges, feature dim 128, 4 cols per thread
__global__ void k_scatter_add(const float* __restrict__ h, const int* __restrict__ src,
                              const int* __restrict__ dst, float* __restrict__ m, int E) {
  long id = (long)blockIdx.x * 256 + threadIdx.x;
  if (id >= (long)E * 32) return;
  int e = (int)(id >> 5);
  int c = (int)(id & 31) << 2;
  int s = src[e], d = dst[e];
  const float4 v = *reinterpret_cast<const float4*>(h + (size_t)s * 128 + c);
  float* p = m + (size_t)d * 128 + c;
  __hip_atomic_fetch_add(p + 0, v.x, __ATOMIC_RELAXED, __HIP_MEMORY_SCOPE_AGENT);
  __hip_atomic_fetch_add(p + 1, v.y, __ATOMIC_RELAXED, __HIP_MEMORY_SCOPE_AGENT);
  __hip_atomic_fetch_add(p + 2, v.z, __ATOMIC_RELAXED, __HIP_MEMORY_SCOPE_AGENT);
  __hip_atomic_fetch_add(p + 3, v.w, __ATOMIC_RELAXED, __HIP_MEMORY_SCOPE_AGENT);
}

// LayerNorm(D=256) + ReLU, writes bf16 for the next WMMA GEMM. One block/row.
__global__ __launch_bounds__(256)
void k_ln_relu_b16(const float* __restrict__ x, const float* __restrict__ g,
                   const float* __restrict__ be, __bf16* __restrict__ o, int rows) {
  __shared__ float red[256];
  int row = blockIdx.x, t = threadIdx.x;
  float v = x[(size_t)row * 256 + t];
  red[t] = v; __syncthreads();
  for (int s = 128; s > 0; s >>= 1) { if (t < s) red[t] += red[t + s]; __syncthreads(); }
  float mu = red[0] * (1.0f / 256.0f); __syncthreads();
  float d = v - mu;
  red[t] = d * d; __syncthreads();
  for (int s = 128; s > 0; s >>= 1) { if (t < s) red[t] += red[t + s]; __syncthreads(); }
  float var = red[0] * (1.0f / 256.0f);
  float y = d * rsqrtf(var + 1e-5f) * g[t] + be[t];
  y = y > 0.f ? y : 0.f;
  o[(size_t)row * 256 + t] = f2bf(y);
}

// ---------------- WMMA GEMM ------------------------------------------------
// C[M,N] = act( A[M,K](bf16) @ Wt[N,K]^T(bf16) + bias + addin ), fp32 accum.
// Block: 8 waves, 128 x (16*NT) tile. Wave: 16 rows x NT WMMA subtiles.
// M must be a multiple of 16 (strip-granular guard keeps EXEC full for WMMA).
// NT is compile-time: the K-loop body is branch-free (loads batched, then
// NT back-to-back v_wmma ops).
union FragU { v16bf v; u32x4 q[2]; };

template <int NT>
__global__ __launch_bounds__(256)
void k_gemm_wmma(const __bf16* __restrict__ A, const __bf16* __restrict__ Wt,
                 const float* __restrict__ bias, const float* __restrict__ addin,
                 float* __restrict__ C, int ldc,
                 __bf16* __restrict__ Cb, int ldcb,
                 int M, int N, int K, int act) {
  const int lane = threadIdx.x & 31;
  const int wave = threadIdx.x >> 5;
  const int strip = blockIdx.x * 128 + wave * 16;
  if (strip >= M) return;                      // wave-uniform: EXEC stays full
  const int n0 = blockIdx.y * (16 * NT);
  const int h  = lane >> 4;                    // lane half: 0 or 1
  const int lm = lane & 15;

  v8f acc[NT];
  #pragma unroll
  for (int t = 0; t < NT; ++t)
    #pragma unroll
    for (int r = 0; r < 8; ++r) acc[t][r] = 0.f;

  // A row pointer (this lane's row, K-contiguous) and per-subtile B pointers.
  const __bf16* aRow = A + (size_t)(strip + lm) * K;
  const __bf16* bCol[NT];
  #pragma unroll
  for (int t = 0; t < NT; ++t)
    bCol[t] = Wt + (size_t)(n0 + t * 16 + lm) * K + h * 16;

  for (int k0 = 0; k0 < K; k0 += 32) {
    FragU af, bw[NT];
    // A 16x32 bf16 fragment: K = h*16+[0..15] contiguous (matches B)
    af.q[0] = __builtin_nontemporal_load(
        reinterpret_cast<const u32x4*>(aRow + k0 + h * 16));
    af.q[1] = __builtin_nontemporal_load(
        reinterpret_cast<const u32x4*>(aRow + k0 + h * 16 + 8));
    // B 32x16 fragments: lane lm = column, K = h*16+[0..15] contiguous in Wt
    #pragma unroll
    for (int t = 0; t < NT; ++t) {
      bw[t].q[0] = *reinterpret_cast<const u32x4*>(bCol[t] + k0);
      bw[t].q[1] = *reinterpret_cast<const u32x4*>(bCol[t] + k0 + 8);
    }
    #pragma unroll
    for (int t = 0; t < NT; ++t)
      acc[t] = __builtin_amdgcn_wmma_f32_16x16x32_bf16(
          false, af.v, false, bw[t].v, (short)0, acc[t], false, false);
  }

  // Epilogue: bias + residual + activation, fp32 store + optional bf16 store.
  #pragma unroll
  for (int t = 0; t < NT; ++t) {
    int col = n0 + t * 16 + lm;
    float bv = bias ? bias[col] : 0.f;
    #pragma unroll
    for (int r = 0; r < 8; ++r) {
      int row = strip + r + h * 8;             // C layout: VGPR r, lane half h
      size_t idx = (size_t)row * ldc + col;
      float v = acc[t][r] + bv;
      if (addin) v += addin[idx];
      if (act == 1)      v = v > 0.f ? v : 0.f;                 // relu
      else if (act == 2) v = v / (1.f + __expf(-v));            // silu
      if (C)  __builtin_nontemporal_store(v, C + idx);
      if (Cb) Cb[(size_t)row * ldcb + col] = f2bf(v);
    }
  }
}

// ---------------- input index map (JAX pytree: dicts sorted by key) --------
enum {
  IN_XU = 0, IN_XI, IN_POS, IN_EUI_S, IN_EUI_D, IN_EIU_S, IN_EIU_D,
  IN_C0_IU_W1, IN_C0_IU_W2, IN_C0_IU_B1, IN_C0_IU_B2,
  IN_C0_UI_W1, IN_C0_UI_W2, IN_C0_UI_B1, IN_C0_UI_B2,
  IN_C1_IU_W1, IN_C1_IU_W2, IN_C1_IU_B1, IN_C1_IU_B2,
  IN_C1_UI_W1, IN_C1_UI_W2, IN_C1_UI_B1, IN_C1_UI_B2,
  IN_HI_W1, IN_HI_W2, IN_HI_W3, IN_HI_B1, IN_HI_B2, IN_HI_B3,
  IN_HI_BE1, IN_HI_BE2, IN_HI_G1, IN_HI_G2,
  IN_HU_W1, IN_HU_W2, IN_HU_W3, IN_HU_B1, IN_HU_B2, IN_HU_B3,
  IN_HU_BE1, IN_HU_BE2, IN_HU_G1, IN_HU_G2,
  IN_JK_I_W, IN_JK_I_B, IN_JK_U_W, IN_JK_U_B,
  IN_PM_W1, IN_PM_W2, IN_PM_B1, IN_PM_B2,
  IN_PR_I_W, IN_PR_I_B, IN_PR_U_W, IN_PR_U_B
};

// ---------------- workspace layout (bytes) ---------------------------------
#define OFF_HU0  ((size_t)0)
#define OFF_HU1  ((size_t)51200000)
#define OFF_HI0  ((size_t)102400000)
#define OFF_HI1  ((size_t)128000000)
#define OFF_MU   ((size_t)153600000)
#define OFF_MI   ((size_t)204800000)
#define OFF_T256 ((size_t)230400000)
#define OFF_B16A ((size_t)332800000)
#define OFF_B16B ((size_t)384000000)
#define OFF_CATU ((size_t)435200000)
#define OFF_CATI ((size_t)486400000)
#define OFF_WTS  ((size_t)512000000)

extern "C" void kernel_launch(void* const* d_in, const int* in_sizes, int n_in,
                              void* d_out, int out_size, void* d_ws, size_t ws_size,
                              hipStream_t stream) {
  (void)in_sizes; (void)n_in; (void)out_size; (void)ws_size;

  char* ws = (char*)d_ws;
  float*  hu0  = (float*)(ws + OFF_HU0);
  float*  hu1  = (float*)(ws + OFF_HU1);
  float*  hi0  = (float*)(ws + OFF_HI0);
  float*  hi1  = (float*)(ws + OFF_HI1);
  float*  m_u  = (float*)(ws + OFF_MU);
  float*  m_i  = (float*)(ws + OFF_MI);
  float*  t256 = (float*)(ws + OFF_T256);
  __bf16* b16A = (__bf16*)(ws + OFF_B16A);
  __bf16* b16B = (__bf16*)(ws + OFF_B16B);
  __bf16* catU = (__bf16*)(ws + OFF_CATU);
  __bf16* catI = (__bf16*)(ws + OFF_CATI);
  __bf16* wts  = (__bf16*)(ws + OFF_WTS);

  const float* pos   = (const float*)d_in[IN_POS];
  const int* eui_s = (const int*)d_in[IN_EUI_S];
  const int* eui_d = (const int*)d_in[IN_EUI_D];
  const int* eiu_s = (const int*)d_in[IN_EIU_S];
  const int* eiu_d = (const int*)d_in[IN_EIU_D];
  #define FP(i) ((const float*)d_in[i])

  // ---- transpose+convert all weight matrices to bf16 [N,K] ----------------
  size_t woff = 0;
  auto tr = [&](int idx, int K, int N) -> __bf16* {
    __bf16* p = wts + woff;
    woff += (size_t)K * N;
    long n = (long)K * N;
    k_transpose_b16<<<dim3((unsigned)((n + 255) / 256)), 256, 0, stream>>>(FP(idx), p, K, N);
    return p;
  };
  __bf16* W_pr_u = tr(IN_PR_U_W, 64, 128);
  __bf16* W_pr_i = tr(IN_PR_I_W, 32, 128);
  __bf16* W_pm1  = tr(IN_PM_W1, 128, 128);
  __bf16* W_pm2  = tr(IN_PM_W2, 128, 128);
  __bf16* WcUI1[2] = { tr(IN_C0_UI_W1, 128, 128), tr(IN_C1_UI_W1, 128, 128) };
  __bf16* WcUI2[2] = { tr(IN_C0_UI_W2, 128, 128), tr(IN_C1_UI_W2, 128, 128) };
  __bf16* WcIU1[2] = { tr(IN_C0_IU_W1, 128, 128), tr(IN_C1_IU_W1, 128, 128) };
  __bf16* WcIU2[2] = { tr(IN_C0_IU_W2, 128, 128), tr(IN_C1_IU_W2, 128, 128) };
  __bf16* W_jk_u = tr(IN_JK_U_W, 256, 128);
  __bf16* W_jk_i = tr(IN_JK_I_W, 256, 128);
  __bf16* W_hu1 = tr(IN_HU_W1, 128, 256);
  __bf16* W_hu2 = tr(IN_HU_W2, 256, 256);
  __bf16* W_hu3 = tr(IN_HU_W3, 256, 32);
  __bf16* W_hi1 = tr(IN_HI_W1, 128, 256);
  __bf16* W_hi2 = tr(IN_HI_W2, 256, 256);
  __bf16* W_hi3 = tr(IN_HI_W3, 256, 16);

  const float* cUIb1[2] = { FP(IN_C0_UI_B1), FP(IN_C1_UI_B1) };
  const float* cUIb2[2] = { FP(IN_C0_UI_B2), FP(IN_C1_UI_B2) };
  const float* cIUb1[2] = { FP(IN_C0_IU_B1), FP(IN_C1_IU_B1) };
  const float* cIUb2[2] = { FP(IN_C0_IU_B2), FP(IN_C1_IU_B2) };

  // NT dispatch: N in {128,256} -> 4 subtiles/wave; N==32 -> 2; N==16 -> 1.
  auto gemm = [&](const __bf16* A, const __bf16* Wt, const float* bias,
                  const float* addin, float* C, int ldc, __bf16* Cb, int ldcb,
                  int M, int N, int K, int act) {
    dim3 g((unsigned)((M + 127) / 128), (unsigned)((N + 63) / 64));
    if ((N & 63) == 0)
      k_gemm_wmma<4><<<g, 256, 0, stream>>>(A, Wt, bias, addin, C, ldc, Cb, ldcb, M, N, K, act);
    else if (N == 32)
      k_gemm_wmma<2><<<g, 256, 0, stream>>>(A, Wt, bias, addin, C, ldc, Cb, ldcb, M, N, K, act);
    else
      k_gemm_wmma<1><<<g, 256, 0, stream>>>(A, Wt, bias, addin, C, ldc, Cb, ldcb, M, N, K, act);
  };
  auto grid1 = [](long n) { return dim3((unsigned)((n + 255) / 256)); };

  // ---- 1) per-type projection --------------------------------------------
  k_convert_b16<<<grid1((long)NU * 64), 256, 0, stream>>>(FP(IN_XU), b16A, (long)NU * 64);
  gemm(b16A, W_pr_u, FP(IN_PR_U_B), nullptr, hu0, 128, nullptr, 0, NU, 128, 64, 0);
  k_convert_b16<<<grid1((long)NI * 32), 256, 0, stream>>>(FP(IN_XI), b16A, (long)NI * 32);
  gemm(b16A, W_pr_i, FP(IN_PR_I_B), nullptr, hi0, 128, nullptr, 0, NI, 128, 32, 0);

  // ---- 2) positional embedding MLP, added into hu ------------------------
  k_posemb<<<grid1((long)NU * 128), 256, 0, stream>>>(pos, b16A, NU);
  gemm(b16A, W_pm1, FP(IN_PM_B1), nullptr, nullptr, 128, b16B, 128, NU, 128, 128, 2); // silu
  gemm(b16B, W_pm2, FP(IN_PM_B2), hu0, hu0, 128, nullptr, 0, NU, 128, 128, 0);        // hu += pe

  // ---- 3) GIN layers, jk='cat' -------------------------------------------
  float* hu_cur = hu0; float* hu_nxt = hu1;
  float* hi_cur = hi0; float* hi_nxt = hi1;
  for (int l = 0; l < 2; ++l) {
    // user -> item
    k_zero<<<grid1((long)NI * 128), 256, 0, stream>>>(m_i, (long)NI * 128);
    k_scatter_add<<<grid1((long)NEDGE * 32), 256, 0, stream>>>(hu_cur, eui_s, eui_d, m_i, NEDGE);
    k_add_convert_b16<<<grid1((long)NI * 128), 256, 0, stream>>>(m_i, hi_cur, b16A, (long)NI * 128);
    gemm(b16A, WcUI1[l], cUIb1[l], nullptr, nullptr, 128, b16B, 128, NI, 128, 128, 1);
    gemm(b16B, WcUI2[l], cUIb2[l], nullptr, hi_nxt, 128, catI + l * 128, 256, NI, 128, 128, 1);
    // item -> user
    k_zero<<<grid1((long)NU * 128), 256, 0, stream>>>(m_u, (long)NU * 128);
    k_scatter_add<<<grid1((long)NEDGE * 32), 256, 0, stream>>>(hi_cur, eiu_s, eiu_d, m_u, NEDGE);
    k_add_convert_b16<<<grid1((long)NU * 128), 256, 0, stream>>>(m_u, hu_cur, b16A, (long)NU * 128);
    gemm(b16A, WcIU1[l], cIUb1[l], nullptr, nullptr, 128, b16B, 128, NU, 128, 128, 1);
    gemm(b16B, WcIU2[l], cIUb2[l], nullptr, hu_nxt, 128, catU + l * 128, 256, NU, 128, 128, 1);
    float* t;
    t = hu_cur; hu_cur = hu_nxt; hu_nxt = t;
    t = hi_cur; hi_cur = hi_nxt; hi_nxt = t;
  }

  // ---- 4) jk linear + MLP heads ------------------------------------------
  float* ju = (float*)d_out;
  float* ji = ju + (size_t)NU * 128;
  float* ou = ji + (size_t)NI * 128;
  float* oi = ou + (size_t)NU * 32;

  // user: jk then head (ju_bf16 lives in b16A until head W1 consumes it)
  gemm(catU, W_jk_u, FP(IN_JK_U_B), nullptr, ju, 128, b16A, 128, NU, 128, 256, 0);
  gemm(b16A, W_hu1, FP(IN_HU_B1), nullptr, t256, 256, nullptr, 0, NU, 256, 128, 0);
  k_ln_relu_b16<<<NU, 256, 0, stream>>>(t256, FP(IN_HU_G1), FP(IN_HU_BE1), b16B, NU);
  gemm(b16B, W_hu2, FP(IN_HU_B2), nullptr, t256, 256, nullptr, 0, NU, 256, 256, 0);
  k_ln_relu_b16<<<NU, 256, 0, stream>>>(t256, FP(IN_HU_G2), FP(IN_HU_BE2), b16A, NU);
  gemm(b16A, W_hu3, FP(IN_HU_B3), nullptr, ou, 32, nullptr, 0, NU, 32, 256, 0);

  // item
  gemm(catI, W_jk_i, FP(IN_JK_I_B), nullptr, ji, 128, b16A, 128, NI, 128, 256, 0);
  gemm(b16A, W_hi1, FP(IN_HI_B1), nullptr, t256, 256, nullptr, 0, NI, 256, 128, 0);
  k_ln_relu_b16<<<NI, 256, 0, stream>>>(t256, FP(IN_HI_G1), FP(IN_HI_BE1), b16B, NI);
  gemm(b16B, W_hi2, FP(IN_HI_B2), nullptr, t256, 256, nullptr, 0, NI, 256, 256, 0);
  k_ln_relu_b16<<<NI, 256, 0, stream>>>(t256, FP(IN_HI_G2), FP(IN_HI_BE2), b16A, NI);
  gemm(b16A, W_hi3, FP(IN_HI_B3), nullptr, oi, 16, nullptr, 0, NI, 16, 256, 0);

  #undef FP
}
